// AtnConv_76149770158348
// MI455X (gfx1250) — matrix operand order repaired and checked by reading, not compiled, not run on hardware
//
#include <hip/hip_runtime.h>

typedef __attribute__((ext_vector_type(16))) _Float16 v16h;
typedef __attribute__((ext_vector_type(8)))  _Float16 v8h;
typedef __attribute__((ext_vector_type(8)))  float    v8f;
typedef __attribute__((ext_vector_type(4)))  unsigned u32x4;
typedef __attribute__((ext_vector_type(8)))  int      i32x8;
typedef __attribute__((ext_vector_type(4)))  int      i32x4;

#define L_      4096      // 64*64 patches / positions
#define KPATCH  576       // C2*3*3
#define CK_     2048      // C1*16 taps
#define C1_     128

// ---- generic f16 GEMM: C[M,N] = A[M,K] x Bt[N,K]^T, f32 accum ----
// LDS A rows stored plain so a lane's A fragment (K ascending within half)
// is one contiguous 32B read; Bt rows stored plain so a lane's B fragment
// (K ascending within half) is one contiguous 32B read.
__global__ __launch_bounds__(256) void k_gemm_f16(
    const _Float16* __restrict__ A, const _Float16* __restrict__ Bt,
    float* __restrict__ C, int M, int N, int K) {
  __shared__ alignas(32) _Float16 sA[64 * 32];
  __shared__ alignas(32) _Float16 sB[64 * 32];
  const int t = threadIdx.x;
  const int wave = t >> 5, lane = t & 31;
  const int wm = wave >> 1, wn = wave & 1;
  const int lm = lane & 15, lh = lane >> 4;
  const int m0 = blockIdx.y * 64, n0 = blockIdx.x * 64;
  const int r8 = t >> 2;                   // tile row 0..63
  const int g8 = t & 3;                    // 8-element K group
  const int pg = (g8 == 1) ? 2 : (g8 == 2) ? 1 : g8;   // middle-group swap
  v8f acc0 = {}, acc1 = {};
  for (int k0 = 0; k0 < K; k0 += 32) {
    __syncthreads();
    *(int4*)&sA[r8 * 32 + g8 * 8] = *(const int4*)&A[(size_t)(m0 + r8) * K + k0 + g8 * 8];
    *(int4*)&sB[r8 * 32 + g8 * 8] = *(const int4*)&Bt[(size_t)(n0 + r8) * K + k0 + g8 * 8];
    if (k0 + 32 < K) {
      __builtin_prefetch(&A[(size_t)(m0 + r8) * K + k0 + 32 + g8 * 8], 0, 1);
      __builtin_prefetch(&Bt[(size_t)(n0 + r8) * K + k0 + 32 + g8 * 8], 0, 1);
    }
    __syncthreads();
    v16h a  = *(const v16h*)&sA[(wm * 16 + lm) * 32 + lh * 16];
    v16h b0 = *(const v16h*)&sB[(wn * 32 + lm) * 32 + lh * 16];
    v16h b1 = *(const v16h*)&sB[(wn * 32 + 16 + lm) * 32 + lh * 16];
    acc0 = __builtin_amdgcn_wmma_f32_16x16x32_f16(false, a, false, b0, (short)0, acc0, false, false);
    acc1 = __builtin_amdgcn_wmma_f32_16x16x32_f16(false, a, false, b1, (short)0, acc1, false, false);
  }
#pragma unroll
  for (int j = 0; j < 8; ++j) {
    int m = m0 + wm * 16 + j + 8 * lh;
    C[(size_t)m * N + n0 + wn * 32 + lm]      = acc0[j];
    C[(size_t)m * N + n0 + wn * 32 + 16 + lm] = acc1[j];
  }
}

// ---- im2col of x2 (3x3, pad1): Xct[p][k] f16 (used as both A and Bt of Gram) ----
__global__ void k_build_xcol(const float* __restrict__ x2, _Float16* __restrict__ Xct) {
  int idx = blockIdx.x * blockDim.x + threadIdx.x;
  if (idx >= L_ * KPATCH) return;
  int p = idx / KPATCH, k = idx - p * KPATCH;
  int c = k / 9, tt = k - c * 9, ky = tt / 3, kx = tt - ky * 3;
  int py = p >> 6, px = p & 63;
  int iy = py + ky - 1, ix = px + kx - 1;
  float v = 0.f;
  if ((unsigned)iy < 64u && (unsigned)ix < 64u)
    v = x2[(size_t)c * 4096 + iy * 64 + ix];
  Xct[(size_t)p * KPATCH + k] = (_Float16)v;
}

// ---- per-patch 1/norm and hole mask mm[q] ----
__global__ __launch_bounds__(256) void k_norm_mask(const _Float16* __restrict__ Xct,
    const float* __restrict__ mask, float* __restrict__ rnorm, float* __restrict__ mm) {
  __shared__ float red[256];
  int p = blockIdx.x;
  float s = 0.f;
  for (int k = threadIdx.x; k < KPATCH; k += 256) {
    float v = (float)Xct[(size_t)p * KPATCH + k];
    s += v * v;
  }
  red[threadIdx.x] = s; __syncthreads();
  for (int st = 128; st > 0; st >>= 1) {
    if (threadIdx.x < st) red[threadIdx.x] += red[threadIdx.x + st];
    __syncthreads();
  }
  if (threadIdx.x == 0) {
    rnorm[p] = 1.f / fmaxf(sqrtf(red[0]), 1e-4f);
    int py = p >> 6, px = p & 63;
    float msum = 0.f;
    for (int ky = 0; ky < 4; ++ky)
      for (int kx = 0; kx < 4; ++kx) {
        int iy = 2 * py + ky - 1, ix = 2 * px + kx - 1;
        if ((unsigned)iy < 128u && (unsigned)ix < 128u) msum += mask[iy * 128 + ix];
      }
    mm[p] = (msum == 0.f) ? 1.f : 0.f;
  }
}

// ---- masked scaled softmax over q for each position p; emits SbT[p][q] f16 ----
__global__ __launch_bounds__(256) void k_softmax(const float* __restrict__ G,
    const float* __restrict__ rnorm, const float* __restrict__ mm,
    const float* __restrict__ ma, _Float16* __restrict__ SbT) {
  __shared__ float red[256];
  int p = blockIdx.x;
  float map = ma[p];
  const float* g = G + (size_t)p * L_;
  float vloc[16];
  float vmax = -1e30f;
#pragma unroll
  for (int i = 0; i < 16; ++i) {
    int q = threadIdx.x + i * 256;
    float v = g[q] * rnorm[q] * mm[q] * map * 10.f;
    vloc[i] = v;
    vmax = fmaxf(vmax, v);
  }
  red[threadIdx.x] = vmax; __syncthreads();
  for (int st = 128; st > 0; st >>= 1) {
    if (threadIdx.x < st) red[threadIdx.x] = fmaxf(red[threadIdx.x], red[threadIdx.x + st]);
    __syncthreads();
  }
  vmax = red[0]; __syncthreads();
  float sum = 0.f;
#pragma unroll
  for (int i = 0; i < 16; ++i) { vloc[i] = __expf(vloc[i] - vmax); sum += vloc[i]; }
  red[threadIdx.x] = sum; __syncthreads();
  for (int st = 128; st > 0; st >>= 1) {
    if (threadIdx.x < st) red[threadIdx.x] += red[threadIdx.x + st];
    __syncthreads();
  }
  float rs = 1.f / red[0];
#pragma unroll
  for (int i = 0; i < 16; ++i) {
    int q = threadIdx.x + i * 256;
    float o = fmaxf(vloc[i] * rs * mm[q] * map, 1e-8f);
    SbT[(size_t)p * L_ + q] = (_Float16)o;
  }
}

// ---- x1 4x4/stride2 patches, transposed: rawWT[ck][q] f16 (Bt of GEMM2) ----
__global__ void k_build_rawWT(const float* __restrict__ x1, _Float16* __restrict__ rawWT) {
  int idx = blockIdx.x * blockDim.x + threadIdx.x;
  if (idx >= CK_ * L_) return;
  int ck = idx >> 12, q = idx & 4095;
  int c = ck >> 4, tt = ck & 15, ky = tt >> 2, kx = tt & 3;
  int qy = q >> 6, qx = q & 63;
  int iy = 2 * qy + ky - 1, ix = 2 * qx + kx - 1;
  float v = 0.f;
  if ((unsigned)iy < 128u && (unsigned)ix < 128u)
    v = x1[(size_t)c * 16384 + iy * 128 + ix];
  rawWT[idx] = (_Float16)v;
}

// ---- transposed-conv scatter (col2im): y image f16 ----
__global__ void k_col2im(const float* __restrict__ Cols, _Float16* __restrict__ ybf) {
  int idx = blockIdx.x * blockDim.x + threadIdx.x;
  if (idx >= C1_ * 128 * 128) return;
  int c = idx >> 14, rem = idx & 16383, y = rem >> 7, x = rem & 127;
  float acc = 0.f;
#pragma unroll
  for (int ky = 0; ky < 4; ++ky) {
    int ty = y + 1 - ky;
    if (ty & 1) continue;
    int iy = ty >> 1;
    if ((unsigned)iy >= 64u) continue;
#pragma unroll
    for (int kx = 0; kx < 4; ++kx) {
      int tx = x + 1 - kx;
      if (tx & 1) continue;
      int ix = tx >> 1;
      if ((unsigned)ix >= 64u) continue;
      acc += Cols[(size_t)(iy * 64 + ix) * CK_ + (c << 4) + (ky << 2) + kx];
    }
  }
  ybf[idx] = (_Float16)(acc * 0.25f);
}

__global__ void k_cvt_w(const float* __restrict__ w, _Float16* __restrict__ wb, int n) {
  int i = blockIdx.x * blockDim.x + threadIdx.x;
  if (i < n) wb[i] = (_Float16)w[i];
}

// ---- dilated group convs, implicit GEMM M=16, K=1152; block=(group,row) ----
// Whole 16x1152 weight panel (36KB) DMA'd once to LDS via the Tensor Data
// Mover; per-step im2col B tiles are wave-private (no barriers in K loop).
__global__ __launch_bounds__(256) void k_dilconv(const _Float16* __restrict__ ybf,
    const _Float16* __restrict__ Wb, const float* __restrict__ bias,
    float* __restrict__ out) {
  __shared__ alignas(32) _Float16 sW[16 * 1152];     // full weight panel
  __shared__ alignas(32) _Float16 sB[8][16 * 32];    // per-wave im2col, [n][k]
  const int g = blockIdx.y;
  const int y = blockIdx.x;
  const int r = 1 << g;                              // rates 1,2,4,8
  const int t = threadIdx.x, wave = t >> 5, lane = t & 31;
  const int lm = lane & 15, lh = lane >> 4;
  const int x0 = wave * 16;
  const _Float16* Ag = Wb + (size_t)g * 16 * 1152;

#if __has_builtin(__builtin_amdgcn_tensor_load_to_lds) && __has_builtin(__builtin_amdgcn_s_wait_tensorcnt)
  if (wave == 0) {
    // D# per CDNA5 ISA: 2D tile 1152x16, f16, row stride 1152
    unsigned long long ga = (unsigned long long)(const void*)Ag;
    unsigned ldsOff = (unsigned)(unsigned long long)(const void*)&sW[0];
    u32x4 g0 = { 1u,                                   // count=1, user mode
                 ldsOff,                               // lds_addr
                 (unsigned)ga,                         // global_addr[31:0]
                 (unsigned)(ga >> 32) | (2u << 30) };  // global_addr[56:32] | type=2
    i32x8 g1 = { (int)(1u << 16),          // data_size=1 (2 bytes)
                 (int)(1152u << 16),       // tensor_dim0[15:0] in bits 63:48
                 (int)(16u << 16),         // tensor_dim0[31:16]=0 | tensor_dim1[15:0]
                 (int)(1152u << 16),       // tensor_dim1[31:16]=0 | tile_dim0=1152
                 (int)16,                  // tile_dim1=16 | tile_dim2=0
                 (int)1152,                // tensor_dim0_stride[31:0]
                 0, 0 };                   // stride0[47:32], stride1 unused (2D)
    i32x4 gz4 = { 0, 0, 0, 0 };
    i32x8 gz8 = { 0, 0, 0, 0, 0, 0, 0, 0 };
    __builtin_amdgcn_tensor_load_to_lds(g0, g1, gz4, gz4, gz8, 0);
    __builtin_amdgcn_s_wait_tensorcnt(0);
  }
  __syncthreads();
#else
  for (int i = t; i < 16 * 1152 / 8; i += 256)
    *(int4*)&sW[i * 8] = *(const int4*)&Ag[i * 8];
  __syncthreads();
#endif

  v8f acc = {};
  for (int k0 = 0; k0 < 1152; k0 += 32) {
    // wave-private im2col tile [n][k]; same-wave LDS ops are in-order
    for (int i = lane; i < 512; i += 32) {
      int n = i >> 5, k = i & 31;
      int kk = k0 + k;
      int c = kk / 9, tt = kk - c * 9, ky = tt / 3, kx = tt - ky * 3;
      int iy = y + r * (ky - 1), ix = x0 + n + r * (kx - 1);
      _Float16 v = (_Float16)0.f;
      if ((unsigned)iy < 128u && (unsigned)ix < 128u)
        v = ybf[(size_t)c * 16384 + iy * 128 + ix];
      sB[wave][n * 32 + k] = v;
    }
    // A fragment straight from plain panel layout: two 16B chunks per lane
    v16h a;
    {
      v8h* ap = (v8h*)&a;
      ap[0] = *(const v8h*)&sW[lm * 1152 + k0 + lh * 16];
      ap[1] = *(const v8h*)&sW[lm * 1152 + k0 + lh * 16 + 8];
    }
    v16h b = *(const v16h*)&sB[wave][lm * 32 + lh * 16];
    acc = __builtin_amdgcn_wmma_f32_16x16x32_f16(false, a, false, b, (short)0, acc, false, false);
  }
#pragma unroll
  for (int j = 0; j < 8; ++j) {
    int o = j + 8 * lh;
    float v = acc[j] + bias[g * 16 + o];
    out[(size_t)(g * 16 + o) * 16384 + y * 128 + x0 + lm] = fmaxf(v, 0.f);
  }
}

extern "C" void kernel_launch(void* const* d_in, const int* in_sizes, int n_in,
                              void* d_out, int out_size, void* d_ws, size_t ws_size,
                              hipStream_t stream) {
  (void)in_sizes; (void)n_in; (void)out_size; (void)ws_size;
  const float* x1       = (const float*)d_in[0];
  const float* x2       = (const float*)d_in[1];
  const float* mask     = (const float*)d_in[2];
  const float* mask_all = (const float*)d_in[3];
  const float* conv_w   = (const float*)d_in[4];
  const float* conv_b   = (const float*)d_in[5];
  float* out = (float*)d_out;
  char* ws = (char*)d_ws;

  // workspace layout (bytes, 256-aligned); Cols aliases G (G dead by then)
  _Float16* XCT   = (_Float16*)(ws);                //  4,718,592
  float*    RNORM = (float*)(ws + 4718592);         //     16,384
  float*    MM    = (float*)(ws + 4734976);         //     16,384
  float*    G     = (float*)(ws + 4751360);         // 67,108,864
  _Float16* SBT   = (_Float16*)(ws + 71860224);     // 33,554,432
  _Float16* RAWWT = (_Float16*)(ws + 105414656);    // 16,777,216
  _Float16* YBF   = (_Float16*)(ws + 122191872);    //  4,194,304
  _Float16* WB    = (_Float16*)(ws + 126386176);    //    147,456
  float*    COLS  = G;

  k_cvt_w<<<(73728 + 255) / 256, 256, 0, stream>>>(conv_w, WB, 73728);

  for (int b = 0; b < 4; ++b) {
    const float* x1b   = x1 + (size_t)b * 128 * 128 * 128;
    const float* x2b   = x2 + (size_t)b * 64 * 64 * 64;
    const float* maskb = mask + (size_t)b * 128 * 128;
    const float* mab   = mask_all + (size_t)b * 4096;

    k_build_xcol<<<(L_ * KPATCH) / 256, 256, 0, stream>>>(x2b, XCT);
    k_norm_mask<<<L_, 256, 0, stream>>>(XCT, maskb, RNORM, MM);
    // Gram: G[p][q] = patch_p . patch_q = Xct x Xct^T  (4096 x 4096 x 576)
    k_gemm_f16<<<dim3(4096 / 64, 4096 / 64), 256, 0, stream>>>(XCT, XCT, G, 4096, 4096, 576);
    k_softmax<<<L_, 256, 0, stream>>>(G, RNORM, MM, mab, SBT);
    k_build_rawWT<<<(CK_ * L_) / 256, 256, 0, stream>>>(x1b, RAWWT);
    // col2im GEMM: Cols[p][ck] = sum_q SbT[p][q] * rawWT[ck][q]  (4096 x 2048 x 4096)
    k_gemm_f16<<<dim3(2048 / 64, 4096 / 64), 256, 0, stream>>>(SBT, RAWWT, COLS, 4096, 2048, 4096);
    k_col2im<<<(C1_ * 128 * 128) / 256, 256, 0, stream>>>(COLS, YBF);
    k_dilconv<<<dim3(128, 4), 256, 0, stream>>>(YBF, WB, conv_b, out + (size_t)b * 64 * 16384);
  }
}
